// VarianceAdaptor_42365557407778
// MI455X (gfx1250) — matrix operand order repaired and checked by reading, not compiled, not run on hardware
//
#include <hip/hip_runtime.h>

// ---------------------------------------------------------------------------
// VarianceAdaptor for MI455X (gfx1250): conv1d layers as bf16 WMMA GEMMs.
// K-pipeline enforced with sched_group_barrier so fragment loads for step s+1
// stay in flight while step s WMMAs execute (compiler was undoing the manual
// double buffer otherwise).
// B=32, L=512, C=256, T=2048, N_BINS=256.
// ---------------------------------------------------------------------------

#define B_DIM 32
#define L_DIM 512
#define C_DIM 256
#define T_DIM 2048
#define NBINS 256
#define TILE_M 64
#define ASTRIDE 264   // padded LDS row stride (ushorts) to skew banks

typedef __bf16 bf16_t;
typedef __attribute__((ext_vector_type(8)))  __bf16 v8bf;
typedef __attribute__((ext_vector_type(16))) __bf16 v16bf;
typedef __attribute__((ext_vector_type(8)))  float  v8f;

#if __has_builtin(__builtin_amdgcn_sched_group_barrier)
#define SGB(mask, size) __builtin_amdgcn_sched_group_barrier((mask), (size), 0)
#else
#define SGB(mask, size)
#endif
// scheduler group masks: 0x8 = MFMA/WMMA, 0x20 = VMEM read, 0x100 = DS read
#define SG_WMMA 0x8
#define SG_VMRD 0x20
#define SG_DSRD 0x100

__device__ __forceinline__ unsigned short f2bf(float f) {
  unsigned int u = __float_as_uint(f);
  unsigned int r = u + 0x7FFFu + ((u >> 16) & 1u);   // round-to-nearest-even
  return (unsigned short)(r >> 16);
}

__device__ __forceinline__ v16bf cat8(v8bf lo, v8bf hi) {
  return __builtin_shufflevector(lo, hi,
      0, 1, 2, 3, 4, 5, 6, 7, 8, 9, 10, 11, 12, 13, 14, 15);
}

// ---- weight prep: w[3][cin][cout] f32 -> wt[3][cout][cin] bf16 -------------
__global__ __launch_bounds__(256) void prep_w_kernel(const float* __restrict__ w,
                                                     unsigned short* __restrict__ wt) {
  int i = blockIdx.x * 256 + threadIdx.x;        // 0 .. 3*256*256-1
  int tap = i >> 16;
  int rem = i & 65535;
  int cin = rem >> 8;
  int cout = rem & 255;
  wt[tap * 65536 + cout * 256 + cin] = f2bf(w[i]);
}

// ---- elementwise f32 -> bf16 ----------------------------------------------
__global__ __launch_bounds__(256) void cvt_bf16_kernel(const float* __restrict__ in,
                                                       unsigned short* __restrict__ out) {
  long i = (long)blockIdx.x * 256 + threadIdx.x;
  out[i] = f2bf(in[i]);
}

// ---------------------------------------------------------------------------
// Core: conv1d(K=3, pad=1, 256->256) as WMMA GEMM + bias + relu.
// A: [M][256] bf16, Wt: [3][256][256] bf16 ([tap][cout][cin]).
// Grid.x = M/64, block = 512 (16 waves). Wave w: M-strip (w&3), N-quarter (w>>2).
// ---------------------------------------------------------------------------
__global__ __launch_bounds__(512)
void conv_gemm_bf16_wmma(const unsigned short* __restrict__ A,
                         const unsigned short* __restrict__ Wt,
                         const float* __restrict__ bias,
                         float* __restrict__ Out,
                         int seqLen) {
  __shared__ unsigned short Ash[(TILE_M + 2) * ASTRIDE];

  const int tid  = threadIdx.x;
  const int lane = tid & 31;
  const int wid  = tid >> 5;       // 0..15
  const int mt   = wid & 3;        // 16-row strip
  const int nq   = wid >> 2;       // 64-col quarter

  const int R0 = blockIdx.x * TILE_M;
  const int seq = R0 / seqLen;
  const int t0  = R0 % seqLen;
  const long seqBase = (long)seq * seqLen;

  const int mrow  = lane & 15;
  const int hi16  = lane >> 4;
  const int m_local = mt * 16 + mrow;
  const int kbase = hi16 * 8;      // A-frag K offset (16-bit A 16x32 layout)
  const int krow0 = hi16 * 16;     // unused; B frags take kbase, same k order as A

  // Pre-load bias for this lane's 4 columns (keeps epilogue free of loads so
  // the scheduler's VMEM groups only see fragment loads).
  float bcol[4];
#pragma unroll
  for (int j = 0; j < 4; ++j) bcol[j] = bias[(nq * 4 + j) * 16 + mrow];

  // Stage halo rows [t0-1 .. t0+64] x 256ci into LDS (zero-pad OOB).
  for (int i = tid; i < (TILE_M + 2) * (C_DIM / 8); i += 512) {
    int hr = i >> 5;               // 0..65
    int ch = i & 31;               // 8-ushort chunk
    int t  = t0 - 1 + hr;
    uint4 v = make_uint4(0u, 0u, 0u, 0u);
    if (t >= 0 && t < seqLen) {
      v = *reinterpret_cast<const uint4*>(A + (seqBase + t) * C_DIM + ch * 8);
    }
    *reinterpret_cast<uint4*>(&Ash[hr * ASTRIDE + ch * 8]) = v;
  }
  __builtin_prefetch(Wt + 65536, 0, 1);    // tap1 weights -> global_prefetch_b8
  __builtin_prefetch(Wt + 131072, 0, 1);   // tap2 weights
  __syncthreads();

  v8f acc[4];
  const v8f vzero = {0.f, 0.f, 0.f, 0.f, 0.f, 0.f, 0.f, 0.f};
#pragma unroll
  for (int j = 0; j < 4; ++j) acc[j] = vzero;

  // step s (0..23): tap = s>>3, cin-block base = (s&7)*32
  auto loadB = [&](int s, v16bf b[4]) {
    const int tap = s >> 3;
    const int k0  = ((s & 7) << 5) + kbase;
    const unsigned short* wtap = Wt + tap * (C_DIM * C_DIM) + k0;
#pragma unroll
    for (int j = 0; j < 4; ++j) {
      const unsigned short* bp = wtap + ((nq * 4 + j) * 16 + mrow) * C_DIM;
      b[j] = cat8(*reinterpret_cast<const v8bf*>(bp),
                  *reinterpret_cast<const v8bf*>(bp + 16));
    }
  };
  auto loadA = [&](int s) -> v16bf {
    const int tap = s >> 3;
    const int cb  = (s & 7) << 5;
    const unsigned short* ap = &Ash[(m_local + tap) * ASTRIDE + cb + kbase];
    return cat8(*reinterpret_cast<const v8bf*>(ap),
                *reinterpret_cast<const v8bf*>(ap + 16));
  };

  v16bf b0[4], b1[4];
  v16bf a0, a1;
  loadB(0, b0);
  a0 = loadA(0);
  SGB(SG_VMRD, 8);                 // step-0 B fragments
  SGB(SG_DSRD, 2);                 // step-0 A fragment

#pragma unroll
  for (int s = 0; s < 24; ++s) {
    v16bf* bc = (s & 1) ? b1 : b0;
    v16bf* bn = (s & 1) ? b0 : b1;
    if (s < 23) {                  // issue next-step loads before this step's math
      loadB(s + 1, bn);
      ((s & 1) ? a0 : a1) = loadA(s + 1);
      SGB(SG_VMRD, 8);             // pipeline: loads(s+1) ...
      SGB(SG_DSRD, 2);
    }
    v16bf ac = (s & 1) ? a1 : a0;
#pragma unroll
    for (int j = 0; j < 4; ++j) {
      acc[j] = __builtin_amdgcn_wmma_f32_16x16x32_bf16(
          false, ac, false, bc[j], (short)0, acc[j], false, false);
    }
    SGB(SG_WMMA, 4);               // ... then WMMAs(s)
  }

  // Epilogue: bias + relu, store f32. D layout: VGPR r -> M=r+8*hi16, N=lane&15.
#pragma unroll
  for (int j = 0; j < 4; ++j) {
    const int col = (nq * 4 + j) * 16 + mrow;
#pragma unroll
    for (int r = 0; r < 8; ++r) {
      const int row_local = mt * 16 + r + hi16 * 8;
      float v = acc[j][r] + bcol[j];
      Out[(long)(R0 + row_local) * C_DIM + col] = v > 0.f ? v : 0.f;
    }
  }
}

// ---- LayerNorm(row) fused with bf16 convert (input to next conv) -----------
__global__ __launch_bounds__(256)
void ln_to_bf16_kernel(const float* __restrict__ H, const float* __restrict__ g,
                       const float* __restrict__ be, unsigned short* __restrict__ out) {
  __shared__ float red[256];
  const long row = blockIdx.x;
  const int c = threadIdx.x;
  float v = H[row * C_DIM + c];
  red[c] = v; __syncthreads();
#pragma unroll
  for (int s = 128; s > 0; s >>= 1) { if (c < s) red[c] += red[c + s]; __syncthreads(); }
  float m = red[0] * (1.f / C_DIM); __syncthreads();
  float d = v - m;
  red[c] = d * d; __syncthreads();
#pragma unroll
  for (int s = 128; s > 0; s >>= 1) { if (c < s) red[c] += red[c + s]; __syncthreads(); }
  float rs = rsqrtf(red[0] * (1.f / C_DIM) + 1e-5f);
  out[row * C_DIM + c] = f2bf(d * rs * g[c] + be[c]);
}

// ---- LayerNorm + (h @ wl + bl), relu -> scalar per row ---------------------
__global__ __launch_bounds__(256)
void linear_ln_kernel(const float* __restrict__ H, const float* __restrict__ g,
                      const float* __restrict__ be, const float* __restrict__ wl,
                      const float* __restrict__ bl, float* __restrict__ out) {
  __shared__ float red[256];
  const long row = blockIdx.x;
  const int c = threadIdx.x;
  float v = H[row * C_DIM + c];
  red[c] = v; __syncthreads();
#pragma unroll
  for (int s = 128; s > 0; s >>= 1) { if (c < s) red[c] += red[c + s]; __syncthreads(); }
  float m = red[0] * (1.f / C_DIM); __syncthreads();
  float d = v - m;
  red[c] = d * d; __syncthreads();
#pragma unroll
  for (int s = 128; s > 0; s >>= 1) { if (c < s) red[c] += red[c + s]; __syncthreads(); }
  float rs = rsqrtf(red[0] * (1.f / C_DIM) + 1e-5f); __syncthreads();
  red[c] = (d * rs * g[c] + be[c]) * wl[c]; __syncthreads();
#pragma unroll
  for (int s = 128; s > 0; s >>= 1) { if (c < s) red[c] += red[c + s]; __syncthreads(); }
  if (c == 0) {
    float r = red[0] + bl[0];
    out[row] = r > 0.f ? r : 0.f;
  }
}

// ---- per-batch inclusive scan of durations (L=512) -------------------------
__global__ __launch_bounds__(512)
void cumsum512_kernel(const int* __restrict__ dur, int* __restrict__ cum) {
  __shared__ int s[512];
  const int b = blockIdx.x, t = threadIdx.x;
  s[t] = dur[b * 512 + t]; __syncthreads();
  for (int off = 1; off < 512; off <<= 1) {
    int v = (t >= off) ? s[t - off] : 0; __syncthreads();
    s[t] += v; __syncthreads();
  }
  cum[b * 512 + t] = s[t];
}

// ---- length regulate: mel[b,t,:] = x[b,idx,:]*mask; positions --------------
__global__ __launch_bounds__(256)
void regulate_kernel(const float* __restrict__ x, const int* __restrict__ cum,
                     float* __restrict__ mel, float* __restrict__ pos) {
  const int bt = blockIdx.x;
  const int b = bt >> 11, t = bt & (T_DIM - 1);
  const int* c = cum + b * L_DIM;
  int lo = 0, hi = L_DIM;                         // searchsorted(cum, t, 'right')
  while (lo < hi) { int mid = (lo + hi) >> 1; if (c[mid] <= t) lo = mid + 1; else hi = mid; }
  const float mask = (t < c[L_DIM - 1]) ? 1.f : 0.f;
  const int idx = lo < (L_DIM - 1) ? lo : (L_DIM - 1);
  const int ch = threadIdx.x;
  mel[(long)bt * C_DIM + ch] = x[((long)b * L_DIM + idx) * C_DIM + ch] * mask;
  if (ch == 0) pos[bt] = (float)(t + 1);
}

// ---- bucketize(lp) [== log(exp(lp)-1+1)] + embedding adds ------------------
__global__ __launch_bounds__(256)
void combine_kernel(const float* __restrict__ mel, const float* __restrict__ lp,
                    const float* __restrict__ le, const float* __restrict__ pb,
                    const float* __restrict__ eb, const float* __restrict__ pemb,
                    const float* __restrict__ eemb, float* __restrict__ out) {
  const int bt = blockIdx.x;
  const float vp = lp[bt], ve = le[bt];
  int plo = 0, phi = NBINS;                        // searchsorted(., 'left')
  while (plo < phi) { int mid = (plo + phi) >> 1; if (pb[mid] < vp) plo = mid + 1; else phi = mid; }
  int elo = 0, ehi = NBINS;
  while (elo < ehi) { int mid = (elo + ehi) >> 1; if (eb[mid] < ve) elo = mid + 1; else ehi = mid; }
  const int pi = plo < (NBINS - 1) ? plo : (NBINS - 1);
  const int ei = elo < (NBINS - 1) ? elo : (NBINS - 1);
  const int ch = threadIdx.x;
  out[(long)bt * C_DIM + ch] =
      mel[(long)bt * C_DIM + ch] + pemb[pi * C_DIM + ch] + eemb[ei * C_DIM + ch];
}

// ---------------------------------------------------------------------------
extern "C" void kernel_launch(void* const* d_in, const int* in_sizes, int n_in,
                              void* d_out, int out_size, void* d_ws, size_t ws_size,
                              hipStream_t stream) {
  const float* x   = (const float*)d_in[0];
  const int*   dur = (const int*)d_in[1];
  // d_in[2] = mel_max_length (fixed 2048)
  const float* P[3][10];   // dp, pp, ep: w1 b1 g1 be1 w2 b2 g2 be2 wl bl
  for (int p = 0; p < 3; ++p)
    for (int k = 0; k < 10; ++k) P[p][k] = (const float*)d_in[3 + p * 10 + k];
  const float* pb   = (const float*)d_in[33];
  const float* eb   = (const float*)d_in[34];
  const float* pemb = (const float*)d_in[35];
  const float* eemb = (const float*)d_in[36];

  const long ML = (long)B_DIM * L_DIM;   // 16384
  const long MT = (long)B_DIM * T_DIM;   // 65536

  // workspace layout (bytes, 16-aligned)
  char* ws = (char*)d_ws;
  unsigned short* wt  = (unsigned short*)(ws);                         // 6 * 3*256*256 bf16
  unsigned short* xb  = (unsigned short*)(ws + 2359296);               // MT*C bf16
  float*          Hb  = (float*)(ws + 2359296 + 33554432);             // MT*C f32
  float*          mel = (float*)(ws + 2359296 + 33554432 + 67108864);  // MT*C f32
  int*            cum = (int*)(ws + 2359296 + 33554432 + 2 * 67108864);

  float* out      = (float*)d_out;
  float* out_mel  = out;                                   // B*T*C
  float* out_ldur = out + (long)MT * C_DIM;                // B*L
  float* out_lp   = out_ldur + ML;                         // B*T
  float* out_le   = out_lp + MT;                           // B*T
  float* out_pos  = out_le + MT;                           // B*T

  auto wlayer = [&](int i) { return wt + (long)i * 3 * C_DIM * C_DIM; };

  // 1) transpose+convert all 6 conv weights to bf16 [tap][cout][cin]
  prep_w_kernel<<<768, 256, 0, stream>>>(P[0][0], wlayer(0));
  prep_w_kernel<<<768, 256, 0, stream>>>(P[0][4], wlayer(1));
  prep_w_kernel<<<768, 256, 0, stream>>>(P[1][0], wlayer(2));
  prep_w_kernel<<<768, 256, 0, stream>>>(P[1][4], wlayer(3));
  prep_w_kernel<<<768, 256, 0, stream>>>(P[2][0], wlayer(4));
  prep_w_kernel<<<768, 256, 0, stream>>>(P[2][4], wlayer(5));

  // 2) duration predictor on x [B*L, 256]
  cvt_bf16_kernel<<<ML * C_DIM / 256, 256, 0, stream>>>(x, xb);
  conv_gemm_bf16_wmma<<<ML / TILE_M, 512, 0, stream>>>(xb, wlayer(0), P[0][1], Hb, L_DIM);
  ln_to_bf16_kernel<<<ML, 256, 0, stream>>>(Hb, P[0][2], P[0][3], xb);
  conv_gemm_bf16_wmma<<<ML / TILE_M, 512, 0, stream>>>(xb, wlayer(1), P[0][5], Hb, L_DIM);
  linear_ln_kernel<<<ML, 256, 0, stream>>>(Hb, P[0][6], P[0][7], P[0][8], P[0][9], out_ldur);

  // 3) length regulate -> mel [B*T, 256] + positions
  cumsum512_kernel<<<B_DIM, 512, 0, stream>>>(dur, cum);
  regulate_kernel<<<MT, 256, 0, stream>>>(x, cum, mel, out_pos);

  // 4) pitch predictor on mel
  cvt_bf16_kernel<<<MT * C_DIM / 256, 256, 0, stream>>>(mel, xb);
  conv_gemm_bf16_wmma<<<MT / TILE_M, 512, 0, stream>>>(xb, wlayer(2), P[1][1], Hb, T_DIM);
  ln_to_bf16_kernel<<<MT, 256, 0, stream>>>(Hb, P[1][2], P[1][3], xb);
  conv_gemm_bf16_wmma<<<MT / TILE_M, 512, 0, stream>>>(xb, wlayer(3), P[1][5], Hb, T_DIM);
  linear_ln_kernel<<<MT, 256, 0, stream>>>(Hb, P[1][6], P[1][7], P[1][8], P[1][9], out_lp);

  // 5) energy predictor on mel
  cvt_bf16_kernel<<<MT * C_DIM / 256, 256, 0, stream>>>(mel, xb);
  conv_gemm_bf16_wmma<<<MT / TILE_M, 512, 0, stream>>>(xb, wlayer(4), P[2][1], Hb, T_DIM);
  ln_to_bf16_kernel<<<MT, 256, 0, stream>>>(Hb, P[2][2], P[2][3], xb);
  conv_gemm_bf16_wmma<<<MT / TILE_M, 512, 0, stream>>>(xb, wlayer(5), P[2][5], Hb, T_DIM);
  linear_ln_kernel<<<MT, 256, 0, stream>>>(Hb, P[2][6], P[2][7], P[2][8], P[2][9], out_le);

  // 6) bucketize + embedding add -> mel_out
  combine_kernel<<<MT, 256, 0, stream>>>(mel, out_lp, out_le, pb, eb, pemb, eemb, out_mel);
}
